// SelfAttentionSparse_44186623541821
// MI455X (gfx1250) — matrix operand order repaired and checked by reading, not compiled, not run on hardware
//
#include <hip/hip_runtime.h>
#include <math.h>

// ---------------- problem constants ----------------
#define BATCH 2
#define T     2048
#define E     1024
#define H     16
#define S     64            // head dim
#define BT    (BATCH * T)   // 4096 total tokens
#define SCALE 0.125f        // 1/sqrt(64)

// gfx1250 async global->LDS path via inline asm (ISA: GLOBAL_LOAD_ASYNC_TO_LDS_B128,
// tracked by ASYNCcnt). Falls back to sync LDS copy elsewhere.
#if defined(__gfx1250__) && __has_builtin(__builtin_amdgcn_s_wait_asynccnt)
#define HAVE_ASYNC 1
#else
#define HAVE_ASYNC 0
#endif

// ---------------- WMMA fragment types ----------------
typedef __bf16 bf16;
typedef __attribute__((ext_vector_type(16))) __bf16 v16bf;
typedef __attribute__((ext_vector_type(8)))  __bf16 v8bf;
typedef __attribute__((ext_vector_type(8)))  float  v8f;

union frag16 { v16bf v; v8bf h[2]; };

// =====================================================
// fp32 -> bf16 elementwise convert
// =====================================================
__global__ __launch_bounds__(256) void cvt_x_kernel(const float* __restrict__ in,
                                                    bf16* __restrict__ out, int n) {
    int i = blockIdx.x * 256 + threadIdx.x;
    if (i < n) out[i] = (__bf16)in[i];
}

// weight convert with triangular masking
// mode 0: dense, mode 1: triu (keep e >= o), mode 2: tril (keep e <= o)
__global__ __launch_bounds__(256) void cvt_w_kernel(const float* __restrict__ in,
                                                    bf16* __restrict__ out, int mode) {
    int i = blockIdx.x * 256 + threadIdx.x;          // over E*E
    int o = i >> 10;                                 // row (output feature)
    int e = i & (E - 1);                             // col (input feature)
    float v = in[i];
    if (mode == 1 && e < o) v = 0.0f;                // triu
    if (mode == 2 && e > o) v = 0.0f;                // tril
    out[i] = (__bf16)v;
}

// =====================================================
// WMMA linear:  out = X[M x K] * W[N x K]^T
// block = 256 threads (8 waves); block tile 128 x 128; wave tile 16 x 128.
// W tile (128 cols x 32 K = 8 KB) double-buffered in LDS via async
// global->LDS copies (ASYNCcnt) so all 8 waves share one global read.
// MODE 0: bf16 row-major out [M][N]
// MODE 1: bf16 transposed out [N][M]     (for V)
// MODE 2: f32 row-major out + bias       (final projection)
// =====================================================
template <int MODE>
__global__ __launch_bounds__(256) void gemm_wmma(const bf16* __restrict__ X,
                                                 const bf16* __restrict__ W,
                                                 void* __restrict__ outv,
                                                 const float* __restrict__ bias,
                                                 int Mtot, int N, int K) {
    __shared__ alignas(16) __bf16 wsmem[2][128 * 32];  // [buf][col][k]

    const int tid  = threadIdx.x;
    const int wave = tid >> 5;
    const int lane = tid & 31;
    const int lr   = lane & 15;
    const int half = lane >> 4;
    const int rbase = blockIdx.x * 128 + wave * 16;
    const int cbase = blockIdx.y * 128;

    // cooperative copy: thread moves 32B of the 8 KB tile: col = tid/2, 16-elem chunk
    const int ccol = tid >> 1;
    const int coff = (tid & 1) * 16;

    auto issue_copy = [&](int buf, int k) {
        const bf16* src = W + (size_t)(cbase + ccol) * K + k + coff;
        __bf16*     dst = &wsmem[buf][ccol * 32 + coff];
#if HAVE_ASYNC
        // ISA 15.18.3: LDS address from the VDST VGPR, global address from VADDR,
        //              (same offset applies to both)
        unsigned lds0 = (unsigned)(size_t)(__attribute__((address_space(3))) __bf16*)dst;
        unsigned long long g0 = (unsigned long long)(size_t)src;
        asm volatile("global_load_async_to_lds_b128 %0, %1, off\n\t"
                     "global_load_async_to_lds_b128 %0, %1, off offset:16"
                     :
                     : "v"(lds0), "v"(g0)
                     : "memory");
#else
        *(v8bf*)dst       = *(const v8bf*)src;
        *(v8bf*)(dst + 8) = *(const v8bf*)(src + 8);
#endif
    };

    v8f c[8] = {};
    const bf16* xp = X + (size_t)(rbase + lr) * K;

    issue_copy(0, 0);
    int buf = 0;
    for (int k = 0; k < K; k += 32, buf ^= 1) {
        if (k + 32 < K) {
            issue_copy(buf ^ 1, k + 32);         // prefetch next tile (other buffer)
#if HAVE_ASYNC
            __builtin_amdgcn_s_wait_asynccnt(2); // retire this buffer's 2 copies
#endif
        } else {
#if HAVE_ASYNC
            __builtin_amdgcn_s_wait_asynccnt(0);
#endif
        }
        __syncthreads();                         // tile visible to all 8 waves

        // A fragment (16 tokens x 32 K): per lane chunks {half*8..}, {16+half*8..}
        frag16 a;
        a.h[0] = *(const v8bf*)(xp + k + half * 8);
        a.h[1] = *(const v8bf*)(xp + k + 16 + half * 8);

        const __bf16* wsb = wsmem[buf];
#pragma unroll
        for (int j = 0; j < 8; ++j) {
            // B fragment (32 K x 16 feats) from LDS: lane = feature col
            frag16 bfr;
            bfr.h[0] = *(const v8bf*)&wsb[(j * 16 + lr) * 32 + half * 8];
            bfr.h[1] = *(const v8bf*)&wsb[(j * 16 + lr) * 32 + 16 + half * 8];
            c[j] = __builtin_amdgcn_wmma_f32_16x16x32_bf16(
                false, a.v, false, bfr.v, (short)0, c[j], false, false);
        }
        __syncthreads();   // reads done before next iteration overwrites this buffer
    }

    if (MODE == 0) {
        bf16* out = (bf16*)outv;
#pragma unroll
        for (int j = 0; j < 8; ++j)
#pragma unroll
            for (int r = 0; r < 8; ++r)
                out[(size_t)(rbase + r + half * 8) * N + cbase + j * 16 + lr] =
                    (__bf16)c[j][r];
    } else if (MODE == 1) {
        bf16* out = (bf16*)outv;   // [N][Mtot]
#pragma unroll
        for (int j = 0; j < 8; ++j) {
            v8bf pk;
#pragma unroll
            for (int r = 0; r < 8; ++r) pk[r] = (__bf16)c[j][r];
            *(v8bf*)(out + (size_t)(cbase + j * 16 + lr) * Mtot + rbase + half * 8) = pk;
        }
    } else {
        float* out = (float*)outv;
#pragma unroll
        for (int j = 0; j < 8; ++j) {
            float bv = bias ? bias[cbase + j * 16 + lr] : 0.0f;
#pragma unroll
            for (int r = 0; r < 8; ++r)
                out[(size_t)(rbase + r + half * 8) * N + cbase + j * 16 + lr] =
                    c[j][r] + bv;
        }
    }
}

// =====================================================
// Flash attention, one wave per (b, h, 16-query tile).
// Q,K bf16 row-major [BT][E]; V transposed bf16 [E][BT]; out bf16 [BT][E].
// =====================================================
__global__ __launch_bounds__(32) void attn_wmma(const bf16* __restrict__ Q,
                                                const bf16* __restrict__ Kmat,
                                                const bf16* __restrict__ Vt,
                                                bf16* __restrict__ O) {
    const int bh    = blockIdx.x;        // 0..B*H-1
    const int b     = bh >> 4;           // H == 16
    const int h     = bh & 15;
    const int qbase = blockIdx.y * 16;
    const int lane  = threadIdx.x;
    const int lr    = lane & 15;
    const int half  = lane >> 4;

    __shared__ alignas(16) __bf16 pshare[16 * 32];  // P tile (C-layout -> A-layout)

    // Q fragments: 16 queries x 64 dims = two K=32 A-frags
    const bf16* qp = Q + ((size_t)(b * T + qbase + lr)) * E + h * S;
    frag16 qa[2];
#pragma unroll
    for (int kc = 0; kc < 2; ++kc) {
        qa[kc].h[0] = *(const v8bf*)(qp + kc * 32 + half * 8);
        qa[kc].h[1] = *(const v8bf*)(qp + kc * 32 + 16 + half * 8);
    }

    v8f of[4] = {};           // O accumulator: 16 queries x 64 dims
    float m[8], l[8];
#pragma unroll
    for (int r = 0; r < 8; ++r) { m[r] = -1e30f; l[r] = 0.0f; }

    for (int kb = 0; kb < T; kb += 32) {
        // ---- scores S = Q K^T for two 16-key subtiles ----
        v8f s[2];
#pragma unroll
        for (int j = 0; j < 2; ++j) {
            const bf16* kp = Kmat + ((size_t)(b * T + kb + j * 16 + lr)) * E + h * S;
            v8f c = {};
#pragma unroll
            for (int kc = 0; kc < 2; ++kc) {
                frag16 kf;
                kf.h[0] = *(const v8bf*)(kp + kc * 32 + half * 8);
                kf.h[1] = *(const v8bf*)(kp + kc * 32 + 16 + half * 8);
                c = __builtin_amdgcn_wmma_f32_16x16x32_bf16(
                    false, qa[kc].v, false, kf.v, (short)0, c, false, false);
            }
            s[j] = c;
        }

        // ---- online softmax (row r+half*8 lives in VGPR r across 16 lanes) ----
        float corr[8];
#pragma unroll
        for (int r = 0; r < 8; ++r) {
            float s0 = s[0][r] * SCALE;
            float s1 = s[1][r] * SCALE;
            float mx = fmaxf(s0, s1);
            mx = fmaxf(mx, __shfl_xor(mx, 1));
            mx = fmaxf(mx, __shfl_xor(mx, 2));
            mx = fmaxf(mx, __shfl_xor(mx, 4));
            mx = fmaxf(mx, __shfl_xor(mx, 8));
            float nm = fmaxf(m[r], mx);
            float cr = __expf(m[r] - nm);
            float p0 = __expf(s0 - nm);
            float p1 = __expf(s1 - nm);
            float rs = p0 + p1;
            rs += __shfl_xor(rs, 1);
            rs += __shfl_xor(rs, 2);
            rs += __shfl_xor(rs, 4);
            rs += __shfl_xor(rs, 8);
            l[r] = l[r] * cr + rs;
            m[r] = nm;
            corr[r] = cr;
            int row = r + half * 8;
            pshare[row * 32 + lr]      = (__bf16)p0;
            pshare[row * 32 + 16 + lr] = (__bf16)p1;
        }
#pragma unroll
        for (int r = 0; r < 8; ++r) {
            of[0][r] *= corr[r]; of[1][r] *= corr[r];
            of[2][r] *= corr[r]; of[3][r] *= corr[r];
        }
        __syncthreads();   // order LDS P-tile writes vs transposed reads

        // ---- P A-fragment (16 q x 32 keys) from LDS ----
        frag16 pa;
        pa.h[0] = *(const v8bf*)&pshare[lr * 32 + half * 8];
        pa.h[1] = *(const v8bf*)&pshare[lr * 32 + 16 + half * 8];

        // ---- O += P V  (V^T rows are contiguous keys) ----
#pragma unroll
        for (int f = 0; f < 4; ++f) {
            const bf16* vp = Vt + ((size_t)(h * S + f * 16 + lr)) * BT
                                + b * T + kb + half * 8;
            frag16 vf;
            vf.h[0] = *(const v8bf*)(vp);
            vf.h[1] = *(const v8bf*)(vp + 16);
            of[f] = __builtin_amdgcn_wmma_f32_16x16x32_bf16(
                false, pa.v, false, vf.v, (short)0, of[f], false, false);
        }
        __syncthreads();   // protect pshare before next iteration's writes
    }

    // ---- epilogue: normalize and store bf16 row-major [BT][E] ----
#pragma unroll
    for (int f = 0; f < 4; ++f)
#pragma unroll
        for (int r = 0; r < 8; ++r) {
            float v = of[f][r] / l[r];
            int row = qbase + r + half * 8;
            O[((size_t)(b * T + row)) * E + h * S + f * 16 + lr] = (__bf16)v;
        }
}

// =====================================================
// host-side launcher
// =====================================================
extern "C" void kernel_launch(void* const* d_in, const int* in_sizes, int n_in,
                              void* d_out, int out_size, void* d_ws, size_t ws_size,
                              hipStream_t stream) {
    const float* x  = (const float*)d_in[0];
    const float* Wk = (const float*)d_in[1];
    const float* Wq = (const float*)d_in[2];
    const float* Wv = (const float*)d_in[3];
    const float* Wu = (const float*)d_in[4];
    const float* bu = (const float*)d_in[5];

    // workspace layout (bf16 elements): 5*BT*E + 4*E*E = 24M elems = 48 MB
    bf16* Xb  = (bf16*)d_ws;
    bf16* Qb  = Xb  + (size_t)BT * E;
    bf16* Kb  = Qb  + (size_t)BT * E;
    bf16* Vt  = Kb  + (size_t)BT * E;   // transposed [E][BT]
    bf16* Ab  = Vt  + (size_t)BT * E;   // attention output [BT][E]
    bf16* Wkb = Ab  + (size_t)BT * E;
    bf16* Wqb = Wkb + (size_t)E * E;
    bf16* Wvb = Wqb + (size_t)E * E;
    bf16* Wub = Wvb + (size_t)E * E;

    // 1) convert inputs to bf16 (mask triangular weights)
    cvt_x_kernel<<<(BT * E) / 256, 256, 0, stream>>>(x, Xb, BT * E);
    cvt_w_kernel<<<(E * E) / 256, 256, 0, stream>>>(Wk, Wkb, 1);  // triu
    cvt_w_kernel<<<(E * E) / 256, 256, 0, stream>>>(Wq, Wqb, 2);  // tril
    cvt_w_kernel<<<(E * E) / 256, 256, 0, stream>>>(Wv, Wvb, 0);
    cvt_w_kernel<<<(E * E) / 256, 256, 0, stream>>>(Wu, Wub, 0);

    // 2) projections (M=4096, N=1024, K=1024); V stored transposed
    dim3 ggrid(BT / 128, E / 128);
    gemm_wmma<0><<<ggrid, 256, 0, stream>>>(Xb, Wkb, Kb, nullptr, BT, E, E);
    gemm_wmma<0><<<ggrid, 256, 0, stream>>>(Xb, Wqb, Qb, nullptr, BT, E, E);
    gemm_wmma<1><<<ggrid, 256, 0, stream>>>(Xb, Wvb, Vt, nullptr, BT, E, E);

    // 3) flash attention: one wave per (b*h, 16-query tile)
    attn_wmma<<<dim3(BATCH * H, T / 16), 32, 0, stream>>>(Qb, Kb, Vt, Ab);

    // 4) final projection with bias -> f32 output
    gemm_wmma<2><<<ggrid, 256, 0, stream>>>(Ab, Wub, d_out, bu, BT, E, E);
}
